// AdaptiveNRI_29703993819981
// MI455X (gfx1250) — matrix operand order repaired and checked by reading, not compiled, not run on hardware
//
#include <hip/hip_runtime.h>

#define N_NODES 10000
#define C_DIM   128
#define E_EDGES 320000

typedef __attribute__((ext_vector_type(16))) _Float16 v16h;
typedef __attribute__((ext_vector_type(8)))  _Float16 v8h;
typedef __attribute__((ext_vector_type(4)))  _Float16 v4h;
typedef __attribute__((ext_vector_type(8)))  float    v8f;

// ---------------------------------------------------------------- activations
__device__ __forceinline__ float act_elu(float x)  { return x > 0.f ? x : __expf(x) - 1.f; }
__device__ __forceinline__ float act_relu(float x) { return x > 0.f ? x : 0.f; }
__device__ __forceinline__ float act_sig(float x)  { return 1.f / (1.f + __expf(-x)); }
__device__ __forceinline__ float apply_act(float x, int act) {
  if (act == 1) return act_elu(x);
  if (act == 2) return act_relu(x);
  if (act == 3) return act_sig(x);
  return x;
}

__device__ __forceinline__ void atomic_add_f32(float* p, float v) {
#if defined(__has_builtin)
#if __has_builtin(__builtin_amdgcn_global_atomic_fadd_f32)
  __builtin_amdgcn_global_atomic_fadd_f32(p, v);
#else
  atomicAdd(p, v);
#endif
#else
  atomicAdd(p, v);
#endif
}

// -------------------------------------------------------- WMMA fragment loads
// A fragment (16x32 f16): lane l holds row (l&15); K-halves: g=l>>4
//   halves[0..7]  = K[g*8 .. g*8+7],  halves[8..15] = K[16+g*8 .. 16+g*8+7]
__device__ __forceinline__ v16h ld_afrag(const _Float16* arow, int g) {
  v8h lo = *(const v8h*)(arow + g * 8);
  v8h hi = *(const v8h*)(arow + 16 + g * 8);
  v16h r;
#pragma unroll
  for (int i = 0; i < 8; i++) { r[i] = lo[i]; r[8 + i] = hi[i]; }
  return r;
}
// B fragment (32x16 f16) from transposed weights (Nc rows x K cols):
// lane l holds col (l&15); halves[0..7] = K[g*8..g*8+7], halves[8..15] = K[16+g*8..16+g*8+7]
__device__ __forceinline__ v16h ld_bfrag(const _Float16* p) {
  v8h lo = *(const v8h*)(p);
  v8h hi = *(const v8h*)(p + 16);
  v16h r;
#pragma unroll
  for (int i = 0; i < 8; i++) { r[i] = lo[i]; r[8 + i] = hi[i]; }
  return r;
}

// ------------------------------------------------------------ prep kernels
__global__ void zero_f32_kernel(float* p, long long n) {
  long long i = (long long)blockIdx.x * blockDim.x + threadIdx.x;
  if (i < n) p[i] = 0.f;
}

// WT[n*K + k] = (f16) W[k*Nc + n]
__global__ void transpose_f16_kernel(const float* __restrict__ W, _Float16* __restrict__ WT,
                                     int K, int Nc) {
  long long i = (long long)blockIdx.x * blockDim.x + threadIdx.x;
  if (i < (long long)K * Nc) {
    int k = (int)(i / Nc), n = (int)(i % Nc);
    WT[(size_t)n * K + k] = (_Float16)W[(size_t)k * Nc + n];
  }
}

// Fold the duplicated-x structure of w_m1a (512x256):
//   Wd = rows[0:128]+rows[128:256], Ws = rows[256:384]+rows[384:512]; store transposed f16
__global__ void fold_w1_kernel(const float* __restrict__ w, _Float16* __restrict__ WdT,
                               _Float16* __restrict__ WsT) {
  int i = blockIdx.x * blockDim.x + threadIdx.x;  // over 128*256
  if (i < 128 * 256) {
    int k = i >> 8, n = i & 255;
    WdT[n * 128 + k] = (_Float16)(w[k * 256 + n] + w[(k + 128) * 256 + n]);
    WsT[n * 128 + k] = (_Float16)(w[(k + 256) * 256 + n] + w[(k + 384) * 256 + n]);
  }
}

// ------------------------------------------------------------ generic WMMA GEMM
// Out(M,Nc) = act( A(M, lda)[:, a_off : a_off+K] @ WT^T + bias ), WT is (Nc x K) f16.
// Block = 256 threads (8 waves), tile 64 rows x 128 cols, K fully unrolled in steps of 32.
// NO branches in the K-loop: out-of-range column tiles are clamped for loads (Nc%16==0,
// M%16==0) and their WMMA results are discarded at the (exec-masked) store epilogue.
#define GM_TILE 64
#define LDSA_STRIDE 40  // 32 halves + 8 pad (80B rows, 16B aligned)

template <int K>
__global__ __launch_bounds__(256) void gemm_kernel(
    const float* __restrict__ A, int lda, int a_off,
    const _Float16* __restrict__ WT, const float* __restrict__ bias,
    float* __restrict__ Out, int M, int Nc, int act) {
  __shared__ __align__(16) _Float16 sA[GM_TILE * LDSA_STRIDE];

  const int tid  = threadIdx.x;
  const int lane = tid & 31;
  const int wave = __builtin_amdgcn_readfirstlane(tid >> 5);  // SGPR -> scalar guards
  const int g = lane >> 4, cl = lane & 15;
  const int wRow = (wave & 3) * 16;   // 0,16,32,48
  const int wCol = (wave >> 2) * 64;  // 0,64
  const int rowBase = blockIdx.y * GM_TILE;
  const int colBase = blockIdx.x * 128;

  // clamped column tiles for loads (always in range; store decides validity)
  int ctl[4];
#pragma unroll
  for (int t = 0; t < 4; t++) {
    int ct = colBase + wCol + t * 16;
    ctl[t] = ct < Nc ? ct : Nc - 16;
  }
  const _Float16* bptr[4];
#pragma unroll
  for (int t = 0; t < 4; t++) bptr[t] = WT + (size_t)(ctl[t] + cl) * K + g * 8;

  v8f acc[4];
#pragma unroll
  for (int t = 0; t < 4; t++)
#pragma unroll
    for (int i = 0; i < 8; i++) acc[t][i] = 0.f;

  // cooperative A staging: 64 rows x 32 cols f32 -> f16 LDS, 8 elems/thread
  const int sRow = tid >> 2;
  const int sCol = (tid & 3) * 8;
  int gRow = rowBase + sRow;
  if (gRow >= M) gRow = M - 1;  // clamp (results from padded rows are never stored)
  const float* aptr = A + (size_t)gRow * lda + a_off;
  const _Float16* amine = &sA[(wRow + cl) * LDSA_STRIDE];

#pragma unroll
  for (int kb = 0; kb < K; kb += 32) {
    __syncthreads();
    float4 f0 = *(const float4*)(aptr + kb + sCol);
    float4 f1 = *(const float4*)(aptr + kb + sCol + 4);
    if (kb + 32 < K) __builtin_prefetch(aptr + kb + 32 + sCol, 0, 1);
    v8h hh;
    hh[0] = (_Float16)f0.x; hh[1] = (_Float16)f0.y; hh[2] = (_Float16)f0.z; hh[3] = (_Float16)f0.w;
    hh[4] = (_Float16)f1.x; hh[5] = (_Float16)f1.y; hh[6] = (_Float16)f1.z; hh[7] = (_Float16)f1.w;
    *(v8h*)(&sA[sRow * LDSA_STRIDE + sCol]) = hh;
    __syncthreads();

    v16h a = ld_afrag(amine, g);
#pragma unroll
    for (int t = 0; t < 4; t++) {
      v16h b = ld_bfrag(bptr[t] + kb);
      acc[t] = __builtin_amdgcn_wmma_f32_16x16x32_f16(
          false, a, false, b, (short)0, acc[t], false, false);
    }
  }

  if (rowBase + wRow < M) {  // scalar branch (M%16==0 -> whole wave tile valid/invalid)
#pragma unroll
    for (int t = 0; t < 4; t++) {
      int ct = colBase + wCol + t * 16;
      if (ct < Nc) {  // scalar branch (Nc%16==0)
        int col = ct + cl;
        float bv = bias ? bias[col] : 0.f;
#pragma unroll
        for (int r2 = 0; r2 < 8; r2++) {
          int row = rowBase + wRow + r2 + 8 * g;
          Out[(size_t)row * Nc + col] = apply_act(acc[t][r2] + bv, act);
        }
      }
    }
  }
}

// ------------------------------------------------------------ edge kernel
// For a tile of 16 edges: h1 = elu(Pd[dst] + Ps[src])  (bias folded into Pd),
// msg = elu(h1 @ w_m1b + b_m1b) via WMMA, then atomic scatter-add into agg[dst].
__global__ __launch_bounds__(256) void edge_kernel(
    const float* __restrict__ Pd, const float* __restrict__ Ps,
    const int* __restrict__ eidx, const _Float16* __restrict__ WT,
    const float* __restrict__ bias, float* __restrict__ agg) {
  __shared__ __align__(16) _Float16 sA[16 * 264];  // 16 edges x 256 halves (+8 pad)
  __shared__ int sDst[16];

  const int tid = threadIdx.x;
  const long long ebase = (long long)blockIdx.x * 16;

  // phase 1: gather + ELU + f16 into LDS
  {
    int r = tid >> 4;             // edge row 0..15
    int cb = (tid & 15) * 16;     // 16 cols per thread
    long long e = ebase + r;
    int d = eidx[E_EDGES + e];
    int s = eidx[e];
    if ((tid & 15) == 0) sDst[r] = d;
    const float* pd = Pd + (size_t)d * 256 + cb;
    const float* ps = Ps + (size_t)s * 256 + cb;
#pragma unroll
    for (int i = 0; i < 16; i += 4) {
      float4 a = *(const float4*)(pd + i);
      float4 b = *(const float4*)(ps + i);
      v4h h;
      h[0] = (_Float16)act_elu(a.x + b.x);
      h[1] = (_Float16)act_elu(a.y + b.y);
      h[2] = (_Float16)act_elu(a.z + b.z);
      h[3] = (_Float16)act_elu(a.w + b.w);
      *(v4h*)(&sA[r * 264 + cb + i]) = h;
    }
  }
  __syncthreads();

  // phase 2: (16x256) @ (256x256) — wave handles 2 column tiles
  const int lane = tid & 31;
  const int wave = __builtin_amdgcn_readfirstlane(tid >> 5);
  const int g = lane >> 4, cl = lane & 15;
  const _Float16* bp0 = WT + (size_t)(wave * 32 + cl) * 256 + g * 8;
  const _Float16* bp1 = WT + (size_t)(wave * 32 + 16 + cl) * 256 + g * 8;
  v8f acc0 = {}, acc1 = {};
#pragma unroll
  for (int kb = 0; kb < 256; kb += 32) {
    v16h a = ld_afrag(&sA[cl * 264 + kb], g);
    v16h b0 = ld_bfrag(bp0 + kb);
    acc0 = __builtin_amdgcn_wmma_f32_16x16x32_f16(false, a, false, b0, (short)0, acc0, false, false);
    v16h b1 = ld_bfrag(bp1 + kb);
    acc1 = __builtin_amdgcn_wmma_f32_16x16x32_f16(false, a, false, b1, (short)0, acc1, false, false);
  }

  // epilogue: elu(acc + bias) scattered via float atomics
  const int col0 = wave * 32 + cl;
  const int col1 = col0 + 16;
  const float bv0 = bias[col0];
  const float bv1 = bias[col1];
#pragma unroll
  for (int r2 = 0; r2 < 8; r2++) {
    int row = r2 + 8 * g;
    int drow = sDst[row];
    atomic_add_f32(&agg[(size_t)drow * 256 + col0], act_elu(acc0[r2] + bv0));
    atomic_add_f32(&agg[(size_t)drow * 256 + col1], act_elu(acc1[r2] + bv1));
  }
}

// ------------------------------------------------------------ launch
static inline void launch_gemm(const float* A, int lda, int a_off, const _Float16* WT,
                               const float* bias, float* Out, int M, int K, int Nc, int act,
                               hipStream_t s) {
  dim3 grid((Nc + 127) / 128, (M + 63) / 64);
  if (K == 128) {
    gemm_kernel<128><<<grid, 256, 0, s>>>(A, lda, a_off, WT, bias, Out, M, Nc, act);
  } else if (K == 256) {
    gemm_kernel<256><<<grid, 256, 0, s>>>(A, lda, a_off, WT, bias, Out, M, Nc, act);
  } else {
    gemm_kernel<384><<<grid, 256, 0, s>>>(A, lda, a_off, WT, bias, Out, M, Nc, act);
  }
}

extern "C" void kernel_launch(void* const* d_in, const int* in_sizes, int n_in,
                              void* d_out, int out_size, void* d_ws, size_t ws_size,
                              hipStream_t stream) {
  const float* api    = (const float*)d_in[0];
  const int*   eidx   = (const int*)d_in[2];
  const float* w_m1a  = (const float*)d_in[7];
  const float* b_m1a  = (const float*)d_in[8];
  const float* w_m1b  = (const float*)d_in[9];
  const float* b_m1b  = (const float*)d_in[10];
  const float* w_m2a  = (const float*)d_in[11];
  const float* b_m2a  = (const float*)d_in[12];
  const float* w_m2b  = (const float*)d_in[13];
  const float* b_m2b  = (const float*)d_in[14];
  const float* w_ma   = (const float*)d_in[15];
  const float* b_ma   = (const float*)d_in[16];
  const float* w_mb   = (const float*)d_in[17];
  const float* b_mb   = (const float*)d_in[18];
  const float* w_inc1 = (const float*)d_in[19];
  const float* b_inc1 = (const float*)d_in[20];
  const float* w_inc2 = (const float*)d_in[21];
  const float* b_inc2 = (const float*)d_in[22];
  float* out = (float*)d_out;

  char* ws = (char*)d_ws;
  size_t off = 0;
  auto alloc = [&](size_t bytes) -> void* {
    void* p = ws + off;
    off += (bytes + 255) & ~(size_t)255;
    return p;
  };

  float* Pd  = (float*)alloc((size_t)N_NODES * 256 * 4);
  float* Ps  = (float*)alloc((size_t)N_NODES * 256 * 4);
  float* agg = (float*)alloc((size_t)N_NODES * 256 * 4);
  float* hA  = (float*)alloc((size_t)N_NODES * 256 * 4);
  float* hB  = (float*)alloc((size_t)N_NODES * 256 * 4);
  float* t1  = (float*)alloc((size_t)N_NODES * 384 * 4);
  _Float16* WdT   = (_Float16*)alloc((size_t)256 * 128 * 2);
  _Float16* WsT   = (_Float16*)alloc((size_t)256 * 128 * 2);
  _Float16* m1bT  = (_Float16*)alloc((size_t)256 * 256 * 2);
  _Float16* m2aT  = (_Float16*)alloc((size_t)256 * 256 * 2);
  _Float16* m2bT  = (_Float16*)alloc((size_t)256 * 256 * 2);
  _Float16* maT   = (_Float16*)alloc((size_t)256 * 256 * 2);
  _Float16* mbT   = (_Float16*)alloc((size_t)256 * 256 * 2);
  _Float16* inc1T = (_Float16*)alloc((size_t)384 * 128 * 2);
  _Float16* inc2T = (_Float16*)alloc((size_t)10000 * 384 * 2);

  // weight prep (f16, transposed to Nc x K) + zero the segment-sum buffer
  fold_w1_kernel<<<(128 * 256 + 255) / 256, 256, 0, stream>>>(w_m1a, WdT, WsT);
  transpose_f16_kernel<<<(256 * 256 + 255) / 256, 256, 0, stream>>>(w_m1b, m1bT, 256, 256);
  transpose_f16_kernel<<<(256 * 256 + 255) / 256, 256, 0, stream>>>(w_m2a, m2aT, 256, 256);
  transpose_f16_kernel<<<(256 * 256 + 255) / 256, 256, 0, stream>>>(w_m2b, m2bT, 256, 256);
  transpose_f16_kernel<<<(256 * 256 + 255) / 256, 256, 0, stream>>>(w_ma, maT, 256, 256);
  transpose_f16_kernel<<<(256 * 256 + 255) / 256, 256, 0, stream>>>(w_mb, mbT, 256, 256);
  transpose_f16_kernel<<<(128 * 384 + 255) / 256, 256, 0, stream>>>(w_inc1, inc1T, 128, 384);
  transpose_f16_kernel<<<(int)(((size_t)384 * 10000 + 255) / 256), 256, 0, stream>>>(w_inc2, inc2T, 384, 10000);
  zero_f32_kernel<<<(int)(((size_t)N_NODES * 256 + 255) / 256), 256, 0, stream>>>(agg, (long long)N_NODES * 256);

  // node-factored edge-MLP layer 1: Pd = A@Wd + b_m1a, Ps = A@Ws
  launch_gemm(api, C_DIM, 0, WdT, b_m1a, Pd, N_NODES, 128, 256, 0, stream);
  launch_gemm(api, C_DIM, 0, WsT, nullptr, Ps, N_NODES, 128, 256, 0, stream);

  // edge layer 2 + segment_sum scatter
  edge_kernel<<<E_EDGES / 16, 256, 0, stream>>>(Pd, Ps, eidx, m1bT, b_m1b, agg);

  // node MLP chain
  launch_gemm(agg, 256, 0, m2aT, b_m2a, hA, N_NODES, 256, 256, 1, stream);
  launch_gemm(hA, 256, 0, m2bT, b_m2b, hB, N_NODES, 256, 256, 1, stream);
  launch_gemm(hB, 256, 0, maT, b_ma, hA, N_NODES, 256, 256, 1, stream);
  launch_gemm(hA, 256, 0, mbT, b_mb, hB, N_NODES, 256, 256, 1, stream);

  // h[:, -C:] @ w_inc1 -> relu, then @ w_inc2 -> sigmoid (final 10000x10000 output)
  launch_gemm(hB, 256, 128, inc1T, b_inc1, t1, N_NODES, 128, 384, 2, stream);
  launch_gemm(t1, 384, 0, inc2T, b_inc2, out, N_NODES, 384, 10000, 3, stream);
}
